// TinyLLM_25752623907483
// MI455X (gfx1250) — matrix operand order repaired and checked by reading, not compiled, not run on hardware
//
#include <hip/hip_runtime.h>
#include <math.h>

// ---------------- model constants ----------------
constexpr int BB   = 2;
constexpr int TT   = 2048;
constexpr int DD   = 1024;
constexpr int HH   = 16;
constexpr int HDIM = 64;          // DD / HH
constexpr int LL   = 4;
constexpr int DFF  = 4096;        // 4*DD
constexpr int VV   = 32000;
constexpr int BT   = BB * TT;     // 4096 tokens

typedef __bf16 bf16;
typedef __attribute__((ext_vector_type(16))) __bf16 v16bf;
typedef __attribute__((ext_vector_type(8)))  float  v8f;

union AFrag { v16bf v; bf16 e[16]; uint4 q[2]; };
union CFrag { v8f   v; float e[8]; };

// ---------------- f32 -> bf16 conversion, WMMA-B swizzled ----------------
// Weight W[K,N] -> out[((n>>4)*(K>>5) + (k>>5))*512 + lane*16 + (k&15)]
// where lane = ((k>>4)&1)*16 + (n&15). Per 32x16 K-tile x N-tile, each lane's
// 16 elements (K-contiguous) are stored contiguously -> ds_load_b128 x2.
__global__ __launch_bounds__(256) void cvt_swzB_kernel(const float* __restrict__ in,
                                                       bf16* __restrict__ out,
                                                       int K, int N) {
    int i = blockIdx.x * 256 + threadIdx.x;
    if (i >= K * N) return;
    int k = i / N, n = i - k * N;
    int kt = k >> 5, kr = k & 31;
    int nt = n >> 4, nr = n & 15;
    int lane = ((kr >> 4) << 4) + nr;
    int e    = kr & 15;
    out[((size_t)nt * (K >> 5) + kt) * 512 + lane * 16 + e] = (bf16)in[i];
}

// ---------------- token + position embedding ----------------
__global__ __launch_bounds__(256) void embed_kernel(const int* __restrict__ idx,
                                                    const float* __restrict__ tok_emb,
                                                    const float* __restrict__ pos_emb,
                                                    float* __restrict__ x) {
    int i = blockIdx.x * 256 + threadIdx.x;           // over BT*DD
    if (i >= BT * DD) return;
    int bt = i / DD, d = i - bt * DD;
    int t = bt % TT;
    x[i] = tok_emb[(size_t)idx[bt] * DD + d] + pos_emb[(size_t)t * DD + d];
}

// ---------------- layernorm (f32 in, bf16 out) ----------------
__global__ __launch_bounds__(256) void ln_kernel(const float* __restrict__ x,
                                                 const float* __restrict__ s,
                                                 const float* __restrict__ b,
                                                 bf16* __restrict__ out) {
    __shared__ float r1[256], r2[256];
    int bt = blockIdx.x;
    const float* xp = x + (size_t)bt * DD;
    float sum = 0.f, sq = 0.f;
    for (int i = threadIdx.x; i < DD; i += 256) { float v = xp[i]; sum += v; sq += v * v; }
    r1[threadIdx.x] = sum; r2[threadIdx.x] = sq; __syncthreads();
    for (int off = 128; off > 0; off >>= 1) {
        if (threadIdx.x < off) { r1[threadIdx.x] += r1[threadIdx.x + off];
                                 r2[threadIdx.x] += r2[threadIdx.x + off]; }
        __syncthreads();
    }
    float mu  = r1[0] * (1.f / DD);
    float var = r2[0] * (1.f / DD) - mu * mu;
    float rs  = rsqrtf(var + 1e-5f);
    for (int i = threadIdx.x; i < DD; i += 256)
        out[(size_t)bt * DD + i] = (bf16)((xp[i] - mu) * rs * s[i] + b[i]);
}

// ---------------- WMMA bf16 GEMM: out = A[M,K] @ W_swz[K,N] + bias (+resid)(+relu)
// block = 256 thr (8 waves), tile 128x128, K step 32. B weights pre-swizzled.
// Double-buffered LDS: global loads for tile kt+1 issue before the WMMAs of
// tile kt, so the matrix pipe hides global latency; one barrier per K-step.
template<int RELU, int RESID, int OUTBF>
__global__ __launch_bounds__(256) void gemm_bf16_wmma(
    const bf16* __restrict__ A, const bf16* __restrict__ Bw,
    const float* __restrict__ bias, const float* __restrict__ resid,
    float* __restrict__ outF, bf16* __restrict__ outB,
    int M, int N, int K)
{
    __shared__ __align__(16) bf16 As[2 * 128 * 32];   // double-buffered A tiles
    __shared__ __align__(16) bf16 Bs[2 * 8 * 512];    // double-buffered B chunks

    const int tid  = threadIdx.x;
    const int lane = tid & 31;
    const int wave = tid >> 5;
    const int wm   = wave & 3;          // M sub-offset: wm*32
    const int wn   = wave >> 2;         // N sub-offset: wn*64  (4 n-tiles)
    const int blockM = blockIdx.y * 128;
    const int blockN = blockIdx.x * 128;
    const int nTile0 = blockIdx.x * 8;
    const int half = lane >> 4;
    const int mrow = lane & 15;
    const int ncol = lane & 15;
    const int Kt   = K >> 5;            // number of K tiles

    AFrag a[2], b[4];
    CFrag c[2][4];
#pragma unroll
    for (int i = 0; i < 2; ++i)
#pragma unroll
        for (int j = 0; j < 4; ++j)
#pragma unroll
            for (int v = 0; v < 8; ++v) c[i][j].e[v] = 0.f;

    const uint4* __restrict__ ga = reinterpret_cast<const uint4*>(A);
    const uint4* __restrict__ gb = reinterpret_cast<const uint4*>(Bw);

    // per-thread staging indices (constant across K-steps)
    const int arow0 = tid >> 2,          acol0 = tid & 3;          // item tid
    const int arow1 = (tid + 256) >> 2,  acol1 = tid & 3;          // item tid+256
    const int bchk0 = tid >> 6,          boff0 = tid & 63;
    const int bchk1 = (tid + 256) >> 6,  boff1 = tid & 63;

    // ---- prologue: stage tile 0 into buffer 0 ----
    {
        uint4 ta0 = ga[(((size_t)(blockM + arow0) * K) >> 3) + acol0];
        uint4 ta1 = ga[(((size_t)(blockM + arow1) * K) >> 3) + acol1];
        uint4 tb0 = gb[((size_t)(nTile0 + bchk0) * Kt) * 64 + boff0];
        uint4 tb1 = gb[((size_t)(nTile0 + bchk1) * Kt) * 64 + boff1];
        uint4* sa = reinterpret_cast<uint4*>(As);
        uint4* sb = reinterpret_cast<uint4*>(Bs);
        sa[tid] = ta0; sa[tid + 256] = ta1;
        sb[tid] = tb0; sb[tid + 256] = tb1;
    }
    __syncthreads();

    for (int kt = 0; kt < Kt; ++kt) {
        const int buf  = kt & 1;
        const bool more = (kt + 1) < Kt;

        // ---- issue next tile's global loads (latency hidden by WMMAs below)
        uint4 na0, na1, nb0, nb1;
        if (more) {
            const int kn = (kt + 1) << 5;
            na0 = ga[(((size_t)(blockM + arow0) * K + kn) >> 3) + acol0];
            na1 = ga[(((size_t)(blockM + arow1) * K + kn) >> 3) + acol1];
            nb0 = gb[((size_t)(nTile0 + bchk0) * Kt + kt + 1) * 64 + boff0];
            nb1 = gb[((size_t)(nTile0 + bchk1) * Kt + kt + 1) * 64 + boff1];
            if (kt + 2 < Kt) {  // WGP-scope prefetch of tile kt+2
                __builtin_prefetch(A + (size_t)(blockM + arow0) * K + ((kt + 2) << 5), 0, 3);
                __builtin_prefetch(Bw + ((size_t)(nTile0 + (tid >> 5)) * Kt + kt + 2) * 512, 0, 3);
            }
        }

        // ---- compute from current buffer ----
        const uint4* sav = reinterpret_cast<const uint4*>(As) + buf * 512;
        const uint4* sbv = reinterpret_cast<const uint4*>(Bs) + buf * 512;
#pragma unroll
        for (int i = 0; i < 2; ++i) {
            int abase = (wm * 32 + i * 16 + mrow) * 4 + half * 2;   // uint4 units
            a[i].q[0] = sav[abase];
            a[i].q[1] = sav[abase + 1];
        }
#pragma unroll
        for (int j = 0; j < 4; ++j) {
            int bbase = (wn * 4 + j) * 64 + lane * 2;           // uint4 units
            b[j].q[0] = sbv[bbase];
            b[j].q[1] = sbv[bbase + 1];
        }
#pragma unroll
        for (int i = 0; i < 2; ++i)
#pragma unroll
            for (int j = 0; j < 4; ++j)
                c[i][j].v = __builtin_amdgcn_wmma_f32_16x16x32_bf16(
                    false, a[i].v, false, b[j].v, (short)0, c[i][j].v, false, false);

        // ---- store next tile into alternate buffer, single barrier ----
        if (more) {
            uint4* sa = reinterpret_cast<uint4*>(As) + (buf ^ 1) * 512;
            uint4* sb = reinterpret_cast<uint4*>(Bs) + (buf ^ 1) * 512;
            sa[tid] = na0; sa[tid + 256] = na1;
            sb[tid] = nb0; sb[tid + 256] = nb1;
        }
        __syncthreads();
    }

    // epilogue: C layout -> row = base + half*8 + v, col = lane&15
#pragma unroll
    for (int i = 0; i < 2; ++i) {
#pragma unroll
        for (int j = 0; j < 4; ++j) {
            int col   = blockN + wn * 64 + j * 16 + ncol;
            int rbase = blockM + wm * 32 + i * 16 + half * 8;
            float bb = bias[col];
#pragma unroll
            for (int v = 0; v < 8; ++v) {
                size_t o = (size_t)(rbase + v) * N + col;
                float val = c[i][j].e[v] + bb;
                if (RESID) val += resid[o];
                if (RELU)  val = val > 0.f ? val : 0.f;
                if (OUTBF) outB[o] = (bf16)val;
                else       outF[o] = val;
            }
        }
    }
}

// ---------------- split qkv [BT,3D] -> q,k,v [B,H,T,HD] f32 (q pre-scaled) ------
__global__ __launch_bounds__(256) void split_kernel(const float* __restrict__ qkv,
                                                    float* __restrict__ q,
                                                    float* __restrict__ k,
                                                    float* __restrict__ v) {
    int i = blockIdx.x * 256 + threadIdx.x;           // over BT*DD
    if (i >= BT * DD) return;
    int bt = i / DD, d = i - bt * DD;
    int h = d / HDIM, hd = d - h * HDIM;
    int b = bt / TT, t = bt - b * TT;
    size_t src = (size_t)bt * (3 * DD) + d;
    size_t dst = (((size_t)(b * HH + h)) * TT + t) * HDIM + hd;
    q[dst] = qkv[src] * 0.125f;                        // 1/sqrt(64)
    k[dst] = qkv[src + DD];
    v[dst] = qkv[src + 2 * DD];
}

// ---------------- causal flash attention: one wave per query row ----------------
__global__ __launch_bounds__(256) void attn_kernel(const float* __restrict__ q,
                                                   const float* __restrict__ k,
                                                   const float* __restrict__ v,
                                                   bf16* __restrict__ ybf) {
    int wave = threadIdx.x >> 5;
    int lane = threadIdx.x & 31;
    int gw = blockIdx.x * 8 + wave;                    // 0 .. B*H*T-1
    int t = gw % TT; int bh = gw / TT;
    int h = bh % HH; int b = bh / HH;
    const float* qp = q + (((size_t)(b * HH + h)) * TT + t) * HDIM;
    const float* kp = k + ((size_t)(b * HH + h)) * TT * HDIM;
    const float* vp = v + ((size_t)(b * HH + h)) * TT * HDIM;
    float q0 = qp[lane], q1 = qp[lane + 32];
    float m = -3.0e38f, l = 0.f, a0 = 0.f, a1 = 0.f;
    for (int s = 0; s <= t; ++s) {
        float d = q0 * kp[(size_t)s * HDIM + lane] + q1 * kp[(size_t)s * HDIM + lane + 32];
        d += __shfl_xor(d, 16, 32);
        d += __shfl_xor(d, 8, 32);
        d += __shfl_xor(d, 4, 32);
        d += __shfl_xor(d, 2, 32);
        d += __shfl_xor(d, 1, 32);
        float nm = fmaxf(m, d);
        float sc = expf(m - nm);
        float p  = expf(d - nm);
        l  = l * sc + p;
        a0 = a0 * sc + p * vp[(size_t)s * HDIM + lane];
        a1 = a1 * sc + p * vp[(size_t)s * HDIM + lane + 32];
        m = nm;
    }
    float inv = 1.f / l;
    bf16* yp = ybf + ((size_t)(b * TT + t)) * DD + h * HDIM;
    yp[lane]      = (bf16)(a0 * inv);
    yp[lane + 32] = (bf16)(a1 * inv);
}

// ---------------- loss: -mean log_softmax(logits)[target] ----------------------
__global__ void zero_loss_kernel(float* loss) { *loss = 0.f; }

__global__ __launch_bounds__(256) void loss_kernel(const float* __restrict__ logits,
                                                   const int* __restrict__ targets,
                                                   float* __restrict__ loss) {
    __shared__ float red[256];
    int bt = blockIdx.x;
    const float* lp = logits + (size_t)bt * VV;
    float mx = -3.0e38f;
    for (int i = threadIdx.x; i < VV; i += 256) mx = fmaxf(mx, lp[i]);
    red[threadIdx.x] = mx; __syncthreads();
    for (int off = 128; off > 0; off >>= 1) {
        if (threadIdx.x < off) red[threadIdx.x] = fmaxf(red[threadIdx.x], red[threadIdx.x + off]);
        __syncthreads();
    }
    mx = red[0]; __syncthreads();
    float sum = 0.f;
    for (int i = threadIdx.x; i < VV; i += 256) sum += expf(lp[i] - mx);
    red[threadIdx.x] = sum; __syncthreads();
    for (int off = 128; off > 0; off >>= 1) {
        if (threadIdx.x < off) red[threadIdx.x] += red[threadIdx.x + off];
        __syncthreads();
    }
    if (threadIdx.x == 0) {
        float lse = mx + logf(red[0]);
        float lpt = lp[targets[bt]] - lse;
        atomicAdd(loss, -lpt / (float)BT);
    }
}

// ---------------- host-side orchestration ----------------
extern "C" void kernel_launch(void* const* d_in, const int* in_sizes, int n_in,
                              void* d_out, int out_size, void* d_ws, size_t ws_size,
                              hipStream_t stream) {
    const int*   idx     = (const int*)d_in[0];
    const int*   targets = (const int*)d_in[1];
    const float* tok_emb = (const float*)d_in[2];
    const float* pos_emb = (const float*)d_in[3];
    const float* qkv_w   = (const float*)d_in[4];
    const float* qkv_b   = (const float*)d_in[5];
    const float* proj_w  = (const float*)d_in[6];
    const float* proj_b  = (const float*)d_in[7];
    const float* ln1_s   = (const float*)d_in[8];
    const float* ln1_b   = (const float*)d_in[9];
    const float* ln2_s   = (const float*)d_in[10];
    const float* ln2_b   = (const float*)d_in[11];
    const float* ffn_w1  = (const float*)d_in[12];
    const float* ffn_b1  = (const float*)d_in[13];
    const float* ffn_w2  = (const float*)d_in[14];
    const float* ffn_b2  = (const float*)d_in[15];
    const float* lnf_s   = (const float*)d_in[16];
    const float* lnf_b   = (const float*)d_in[17];
    const float* head_w  = (const float*)d_in[18];
    const float* head_b  = (const float*)d_in[19];

    float* logits = (float*)d_out;                       // [BT, VV]
    float* loss   = logits + (size_t)BT * VV;

    char* ws = (char*)d_ws; size_t off = 0;
    auto alloc = [&](size_t bytes) -> void* {
        void* p = ws + off; off = (off + bytes + 255) & ~(size_t)255; return p;
    };
    const size_t nWqkv = (size_t)LL * DD * 3 * DD;
    const size_t nWprj = (size_t)LL * DD * DD;
    const size_t nWf1  = (size_t)LL * DD * DFF;
    const size_t nWf2  = (size_t)LL * DFF * DD;
    const size_t nWhd  = (size_t)DD * VV;

    bf16*  wqkv = (bf16*)alloc(nWqkv * sizeof(bf16));
    bf16*  wprj = (bf16*)alloc(nWprj * sizeof(bf16));
    bf16*  wf1  = (bf16*)alloc(nWf1  * sizeof(bf16));
    bf16*  wf2  = (bf16*)alloc(nWf2  * sizeof(bf16));
    bf16*  whd  = (bf16*)alloc(nWhd  * sizeof(bf16));
    float* x    = (float*)alloc((size_t)BT * DD * sizeof(float));
    bf16*  hbf  = (bf16*) alloc((size_t)BT * DD * sizeof(bf16));
    float* qkvf = (float*)alloc((size_t)BT * 3 * DD * sizeof(float));
    float* qf   = (float*)alloc((size_t)BT * DD * sizeof(float));
    float* kf   = (float*)alloc((size_t)BT * DD * sizeof(float));
    float* vf   = (float*)alloc((size_t)BT * DD * sizeof(float));
    bf16*  ybf  = (bf16*) alloc((size_t)BT * DD * sizeof(bf16));
    bf16*  ffnh = (bf16*) alloc((size_t)BT * DFF * sizeof(bf16));

    // convert + swizzle each weight matrix (per layer) into WMMA-B order
    auto cvtswz = [&](const float* src, bf16* dst, int K, int N) {
        int n = K * N;
        cvt_swzB_kernel<<<dim3((unsigned)((n + 255) / 256)), 256, 0, stream>>>(src, dst, K, N);
    };
    for (int l = 0; l < LL; ++l) {
        cvtswz(qkv_w  + (size_t)l * DD * 3 * DD, wqkv + (size_t)l * DD * 3 * DD, DD, 3 * DD);
        cvtswz(proj_w + (size_t)l * DD * DD,     wprj + (size_t)l * DD * DD,     DD, DD);
        cvtswz(ffn_w1 + (size_t)l * DD * DFF,    wf1  + (size_t)l * DD * DFF,    DD, DFF);
        cvtswz(ffn_w2 + (size_t)l * DFF * DD,    wf2  + (size_t)l * DFF * DD,    DFF, DD);
    }
    cvtswz(head_w, whd, DD, VV);

    embed_kernel<<<(BT * DD) / 256, 256, 0, stream>>>(idx, tok_emb, pos_emb, x);

    for (int l = 0; l < LL; ++l) {
        // attention block
        ln_kernel<<<BT, 256, 0, stream>>>(x, ln1_s + (size_t)l * DD, ln1_b + (size_t)l * DD, hbf);
        gemm_bf16_wmma<0, 0, 0><<<dim3(3 * DD / 128, BT / 128), 256, 0, stream>>>(
            hbf, wqkv + (size_t)l * DD * 3 * DD, qkv_b + (size_t)l * 3 * DD,
            nullptr, qkvf, nullptr, BT, 3 * DD, DD);
        split_kernel<<<(BT * DD) / 256, 256, 0, stream>>>(qkvf, qf, kf, vf);
        attn_kernel<<<BB * HH * TT / 8, 256, 0, stream>>>(qf, kf, vf, ybf);
        gemm_bf16_wmma<0, 1, 0><<<dim3(DD / 128, BT / 128), 256, 0, stream>>>(
            ybf, wprj + (size_t)l * DD * DD, proj_b + (size_t)l * DD,
            x, x, nullptr, BT, DD, DD);
        // FFN block
        ln_kernel<<<BT, 256, 0, stream>>>(x, ln2_s + (size_t)l * DD, ln2_b + (size_t)l * DD, hbf);
        gemm_bf16_wmma<1, 0, 1><<<dim3(DFF / 128, BT / 128), 256, 0, stream>>>(
            hbf, wf1 + (size_t)l * DD * DFF, ffn_b1 + (size_t)l * DFF,
            nullptr, nullptr, ffnh, BT, DFF, DD);
        gemm_bf16_wmma<0, 1, 0><<<dim3(DD / 128, BT / 128), 256, 0, stream>>>(
            ffnh, wf2 + (size_t)l * DFF * DD, ffn_b2 + (size_t)l * DD,
            x, x, nullptr, BT, DD, DFF);
    }

    ln_kernel<<<BT, 256, 0, stream>>>(x, lnf_s, lnf_b, hbf);
    gemm_bf16_wmma<0, 0, 0><<<dim3(VV / 128, BT / 128), 256, 0, stream>>>(
        hbf, whd, head_b, nullptr, logits, nullptr, BT, VV, DD);

    zero_loss_kernel<<<1, 1, 0, stream>>>(loss);
    loss_kernel<<<BT, 256, 0, stream>>>(logits, targets, loss);
}
